// MambaModel_8469675507747
// MI455X (gfx1250) — compile-verified, matrix operand order checked by reading, not run on hardware
//
#include <hip/hip_runtime.h>
#include <cstdint>
#include <cstddef>

typedef __attribute__((ext_vector_type(16))) _Float16 v16h;
typedef __attribute__((ext_vector_type(8)))  _Float16 v8h;
typedef __attribute__((ext_vector_type(8)))  float    v8f;

// Model constants
constexpr int BBATCH = 8;
constexpr int LSEQ   = 1024;
constexpr int PDIM   = 64;
constexpr int HDIM   = 512;
constexpr int NLAYER = 4;
constexpr int DIN    = 1024;   // 2*H
constexpr int NST    = 16;     // state dim
constexpr int DTRANK = 32;
constexpr int DCONVW = 4;
constexpr int BL     = BBATCH * LSEQ;   // 8192 rows

// fast stable softplus: max(x,0) + log(1 + exp(-|x|))
__device__ __forceinline__ float softplus_fast(float x) {
    return fmaxf(x, 0.f) + __logf(1.f + __expf(-fabsf(x)));
}
__device__ __forceinline__ float silu_fast(float x) {
    return x / (1.f + __expf(-x));
}

// ---------------------------------------------------------------------------
// 0) f32 -> f16 convert (weights, once per launch)
// ---------------------------------------------------------------------------
__global__ __launch_bounds__(256)
void cvt_f16_kernel(const float* __restrict__ src, _Float16* __restrict__ dst, int count)
{
    int i = blockIdx.x * blockDim.x + threadIdx.x;
    if (i < count) dst[i] = (_Float16)src[i];
}

// ---------------------------------------------------------------------------
// 1) min-max normalize + LayerNorm over P=64 -> f16.  One wave per row.
// ---------------------------------------------------------------------------
__global__ __launch_bounds__(256)
void prep_kernel(const float* __restrict__ x, const float* __restrict__ g,
                 const float* __restrict__ beta, _Float16* __restrict__ xlh, int rows)
{
    int row  = blockIdx.x * (blockDim.x >> 5) + (threadIdx.x >> 5);
    if (row >= rows) return;
    int lane = threadIdx.x & 31;
    const float* xr = x + (size_t)row * PDIM;
    float a0 = xr[lane], a1 = xr[lane + 32];

    float mn = fminf(a0, a1), mx = fmaxf(a0, a1);
    #pragma unroll
    for (int o = 16; o > 0; o >>= 1) {
        mn = fminf(mn, __shfl_xor(mn, o, 32));
        mx = fmaxf(mx, __shfl_xor(mx, o, 32));
    }
    float inv = 1.f / (mx - mn + 1e-6f);
    float n0 = (a0 - mn) * inv, n1 = (a1 - mn) * inv;

    float s = n0 + n1;
    #pragma unroll
    for (int o = 16; o > 0; o >>= 1) s += __shfl_xor(s, o, 32);
    float mu = s * (1.f / 64.f);
    float d0 = n0 - mu, d1 = n1 - mu;

    float v = d0 * d0 + d1 * d1;
    #pragma unroll
    for (int o = 16; o > 0; o >>= 1) v += __shfl_xor(v, o, 32);
    float rstd = rsqrtf(v * (1.f / 64.f) + 1e-5f);

    _Float16* o = xlh + (size_t)row * PDIM;
    o[lane]      = (_Float16)(d0 * rstd * g[lane]      + beta[lane]);
    o[lane + 32] = (_Float16)(d1 * rstd * g[lane + 32] + beta[lane + 32]);
}

// ---------------------------------------------------------------------------
// 2) WMMA GEMM, all-f16 operands, fp32 accumulate.
//    Block = 8 waves = 128(M) x 64(N) supertile; waves share the N-strip.
//    Per K-chunk(32): cooperative stage of the 64x32 f16 W chunk into LDS
//    (rows padded to 40 halves -> conflict-free ds_load_b128), then each
//    wave: 2 global b128 (A frag) + 8 LDS b128 (4 B frags) + 4 v_wmma.
//    C written as f32 (C) and/or f16 (Ch) with optional bias / softplus.
// ---------------------------------------------------------------------------
__global__ __launch_bounds__(256)
void wmma_gemm_kernel(const _Float16* __restrict__ A, const _Float16* __restrict__ W,
                      const float* __restrict__ bias,
                      float* __restrict__ C, _Float16* __restrict__ Ch,
                      int M, int N, int K, int lda, int act)
{
    constexpr int LROW = 40;                    // padded LDS row (halves)
    __shared__ _Float16 lw[64 * LROW];

    int tilesN = N >> 6;
    int tn     = blockIdx.x % tilesN;
    int tmBase = (blockIdx.x / tilesN) * 8;     // 8 M-tiles per block
    int wv     = threadIdx.x >> 5;
    int tm     = tmBase + wv;

    int lane = threadIdx.x & 31;
    int hi   = lane >> 4;
    int r    = lane & 15;

    // cooperative W loader mapping: 64 rows x 4 segments of 8 halves
    int ldr  = threadIdx.x >> 2;
    int lseg = threadIdx.x & 3;
    const _Float16* wsrc = W + (size_t)(tn * 64 + ldr) * K + lseg * 8;
    _Float16*       ldst = &lw[ldr * LROW + lseg * 8];

    const _Float16* arow = A + (size_t)(tm * 16 + r) * lda + hi * 8;

    v8f acc[4] = {{}, {}, {}, {}};
    for (int k0 = 0; k0 < K; k0 += 32) {
        *(v8h*)ldst = *(const v8h*)(wsrc + k0);         // stage W chunk
        __syncthreads();

        v8h alo = *(const v8h*)(arow + k0);             // K = k0+hi*8 ..
        v8h ahi = *(const v8h*)(arow + k0 + 16);        // K = k0+16+hi*8 ..
        v16h a  = __builtin_shufflevector(alo, ahi,
                      0, 1, 2, 3, 4, 5, 6, 7, 8, 9, 10, 11, 12, 13, 14, 15);

        #pragma unroll
        for (int c = 0; c < 4; ++c) {
            const _Float16* lp = &lw[(c * 16 + r) * LROW + hi * 8];
            v8h blo = *(const v8h*)lp;
            v8h bhi = *(const v8h*)(lp + 16);
            v16h bb = __builtin_shufflevector(blo, bhi,
                          0, 1, 2, 3, 4, 5, 6, 7, 8, 9, 10, 11, 12, 13, 14, 15);
            acc[c] = __builtin_amdgcn_wmma_f32_16x16x32_f16(
                         false, a, false, bb, (short)0, acc[c], false, false);
        }
        __syncthreads();
    }

    #pragma unroll
    for (int c = 0; c < 4; ++c) {
        int   n  = tn * 64 + c * 16 + r;
        float bi = bias ? bias[n] : 0.f;
        #pragma unroll
        for (int i = 0; i < 8; ++i) {
            int   m = tm * 16 + hi * 8 + i;     // C/D layout: M = vgpr + 8*hi
            float v = acc[c][i] + bi;
            if (act == 1) v = softplus_fast(v);
            if (C)  C[(size_t)m * N + n]  = v;
            if (Ch) Ch[(size_t)m * N + n] = (_Float16)v;
        }
    }
}

// ---------------------------------------------------------------------------
// 3) Depthwise causal conv (d_conv=4) + bias + SiLU; writes f32 u (scan)
//    and f16 uh (x_proj A-side).
// ---------------------------------------------------------------------------
__global__ __launch_bounds__(256)
void conv_silu_kernel(const float* __restrict__ xz, const float* __restrict__ cw,
                      const float* __restrict__ cb, float* __restrict__ u,
                      _Float16* __restrict__ uh)
{
    int idx = blockIdx.x * blockDim.x + threadIdx.x;   // over BL*DIN
    int d   = idx & (DIN - 1);
    int bl  = idx >> 10;                                // DIN = 1024
    int l   = bl & (LSEQ - 1);

    float w0 = cw[d * 4 + 0], w1 = cw[d * 4 + 1];
    float w2 = cw[d * 4 + 2], w3 = cw[d * 4 + 3];
    const size_t rs   = 2 * DIN;
    size_t       base = (size_t)bl * rs + d;

    float v = cb[d] + w3 * xz[base];
    if (l >= 1) v += w2 * xz[base - rs];
    if (l >= 2) v += w1 * xz[base - 2 * rs];
    if (l >= 3) v += w0 * xz[base - 3 * rs];
    float su = silu_fast(v);
    u [(size_t)bl * DIN + d] = su;
    uh[(size_t)bl * DIN + d] = (_Float16)su;
}

// ---------------------------------------------------------------------------
// 4) Selective scan, one LANE per (b,d,n): 131072 lanes.  Per step:
//    1 v_exp_f32 + 2 FMA per lane, y reduced over the 16 n-lanes via
//    __shfl_xor (offsets 1/2/4/8 stay inside the group).  Writes f16 y
//    (out_proj A-side).  dt/u are broadcast loads; B/C coalesced over n.
// ---------------------------------------------------------------------------
__global__ __launch_bounds__(256)
void scan_kernel(const float* __restrict__ u, const float* __restrict__ delta,
                 const float* __restrict__ dbl, const float* __restrict__ A_log,
                 const float* __restrict__ Dp, const float* __restrict__ xz,
                 _Float16* __restrict__ yh)
{
    int idx = blockIdx.x * blockDim.x + threadIdx.x;    // over B*DIN*NST
    int n   = idx & (NST - 1);
    int bd  = idx >> 4;
    int d   = bd & (DIN - 1);
    int b   = bd >> 10;

    float Ar = -__expf(A_log[d * NST + n]);
    float Dd = Dp[d];
    float h  = 0.f;

    for (int t = 0; t < LSEQ; ++t) {
        size_t bl = (size_t)b * LSEQ + t;
        float  dt = delta[bl * DIN + d];                // broadcast across n
        float  ut = u[bl * DIN + d];                    // broadcast across n
        const float* bc = dbl + bl * 64;                // [32..47]=B, [48..63]=C
        h = __expf(dt * Ar) * h + (dt * ut) * bc[32 + n];
        float yv = h * bc[48 + n];
        #pragma unroll
        for (int o = 8; o > 0; o >>= 1) yv += __shfl_xor(yv, o, 32);
        if (n == 0) {
            float z = xz[bl * (2 * DIN) + DIN + d];
            yh[bl * DIN + d] = (_Float16)((yv + ut * Dd) * silu_fast(z));
        }
    }
}

// ---------------------------------------------------------------------------
// 5) max-pool over time + FC(512 -> 1).  One block per batch element.
// ---------------------------------------------------------------------------
__global__ __launch_bounds__(256)
void final_kernel(const float* __restrict__ h, const float* __restrict__ fcw,
                  const float* __restrict__ fcb, float* __restrict__ out)
{
    __shared__ float red[256];
    int b = blockIdx.x;
    float acc = 0.f;
    for (int c = threadIdx.x; c < HDIM; c += blockDim.x) {
        float m = -3.4e38f;
        const float* hp = h + (size_t)b * LSEQ * HDIM + c;
        for (int t = 0; t < LSEQ; ++t) m = fmaxf(m, hp[(size_t)t * HDIM]);
        acc += m * fcw[c];
    }
    red[threadIdx.x] = acc;
    __syncthreads();
    for (int s = 128; s > 0; s >>= 1) {
        if (threadIdx.x < s) red[threadIdx.x] += red[threadIdx.x + s];
        __syncthreads();
    }
    if (threadIdx.x == 0) out[b] = red[0] + fcb[0];
}

// ---------------------------------------------------------------------------
static inline void launch_gemm(const _Float16* A, const _Float16* W, const float* bias,
                               float* C, _Float16* Ch,
                               int M, int N, int K, int lda, int act, hipStream_t s)
{
    int blocks = (M >> 7) * (N >> 6);           // 128x64 supertile per block
    wmma_gemm_kernel<<<blocks, 256, 0, s>>>(A, W, bias, C, Ch, M, N, K, lda, act);
}
static inline void launch_cvt(const float* src, _Float16* dst, int count, hipStream_t s)
{
    cvt_f16_kernel<<<(count + 255) / 256, 256, 0, s>>>(src, dst, count);
}

extern "C" void kernel_launch(void* const* d_in, const int* in_sizes, int n_in,
                              void* d_out, int out_size, void* d_ws, size_t ws_size,
                              hipStream_t stream)
{
    const float* x         = (const float*)d_in[0];
    const float* ln_g      = (const float*)d_in[1];
    const float* ln_b      = (const float*)d_in[2];
    const float* proj_w    = (const float*)d_in[3];
    const float* proj_b    = (const float*)d_in[4];
    const float* in_proj_w = (const float*)d_in[5];
    const float* conv_w    = (const float*)d_in[6];
    const float* conv_b    = (const float*)d_in[7];
    const float* xproj_w   = (const float*)d_in[8];
    const float* dtproj_w  = (const float*)d_in[9];
    const float* dtproj_b  = (const float*)d_in[10];
    const float* A_log     = (const float*)d_in[11];
    const float* Dp        = (const float*)d_in[12];
    const float* out_pw    = (const float*)d_in[13];
    const float* fc_w      = (const float*)d_in[14];
    const float* fc_b      = (const float*)d_in[15];
    float* out = (float*)d_out;

    // ---- workspace carve-up -----------------------------------------------
    // f32 region
    float* wsf = (float*)d_ws;
    size_t off = 0;
    float* hbuf  = wsf + off; off += (size_t)BL * HDIM;      // 16 MB
    float* xz    = wsf + off; off += (size_t)BL * 2 * DIN;   // 64 MB
    float* u     = wsf + off; off += (size_t)BL * DIN;       // 32 MB
    float* dblb  = wsf + off; off += (size_t)BL * 64;        //  2 MB
    float* delta = wsf + off; off += (size_t)BL * DIN;       // 32 MB
    // f16 region (starts 16B-aligned: all f32 sizes are multiples of 64)
    _Float16* wsh = (_Float16*)(wsf + off);
    size_t hoff = 0;
    _Float16* xlh   = wsh + hoff; hoff += (size_t)BL * PDIM;   //  1 MB
    _Float16* hbufh = wsh + hoff; hoff += (size_t)BL * HDIM;   //  8 MB
    _Float16* uh    = wsh + hoff; hoff += (size_t)BL * DIN;    // 16 MB
    _Float16* dblh  = wsh + hoff; hoff += (size_t)BL * 64;     //  1 MB
    _Float16* yh    = wsh + hoff; hoff += (size_t)BL * DIN;    // 16 MB
    // f16 weights
    _Float16* proj_wh = wsh + hoff; hoff += (size_t)HDIM * PDIM;
    _Float16* in_wh   = wsh + hoff; hoff += (size_t)NLAYER * 2 * DIN * HDIM;
    _Float16* xp_wh   = wsh + hoff; hoff += (size_t)NLAYER * (DTRANK + 2 * NST) * DIN;
    _Float16* dt_wh   = wsh + hoff; hoff += (size_t)NLAYER * DIN * DTRANK;
    _Float16* op_wh   = wsh + hoff; hoff += (size_t)NLAYER * HDIM * DIN;

    // ---- one-time weight conversion to f16 --------------------------------
    launch_cvt(proj_w,    proj_wh, HDIM * PDIM,                         stream);
    launch_cvt(in_proj_w, in_wh,   NLAYER * 2 * DIN * HDIM,             stream);
    launch_cvt(xproj_w,   xp_wh,   NLAYER * (DTRANK + 2 * NST) * DIN,   stream);
    launch_cvt(dtproj_w,  dt_wh,   NLAYER * DIN * DTRANK,               stream);
    launch_cvt(out_pw,    op_wh,   NLAYER * HDIM * DIN,                 stream);

    // ---- pre: normalize + LN (-> f16), project to H (-> f16) --------------
    prep_kernel<<<BL / 8, 256, 0, stream>>>(x, ln_g, ln_b, xlh, BL);
    launch_gemm(xlh, proj_wh, proj_b, nullptr, hbufh, BL, HDIM, PDIM, PDIM, 0, stream);

    for (int i = 0; i < NLAYER; ++i) {
        // xz = h @ in_proj^T                       (8192 x 2048, f32)
        launch_gemm(hbufh, in_wh + (size_t)i * 2 * DIN * HDIM, nullptr,
                    xz, nullptr, BL, 2 * DIN, HDIM, HDIM, 0, stream);
        // u = silu(causal depthwise conv(xc) + b)  (f32 + f16)
        conv_silu_kernel<<<(BL * DIN) / 256, 256, 0, stream>>>(
            xz, conv_w + (size_t)i * DIN * DCONVW, conv_b + (size_t)i * DIN, u, uh);
        // dbl = u @ x_proj^T                       (8192 x 64: dt|B|C, f32+f16)
        launch_gemm(uh, xp_wh + (size_t)i * 64 * DIN, nullptr,
                    dblb, dblh, BL, 64, DIN, DIN, 0, stream);
        // delta = softplus(dt @ dt_proj^T + b)     (A slice: lda=64, K=32)
        launch_gemm(dblh, dt_wh + (size_t)i * DIN * DTRANK,
                    dtproj_b + (size_t)i * DIN, delta, nullptr,
                    BL, DIN, DTRANK, 64, 1, stream);
        // selective scan + D skip + SiLU(z) gate   -> f16 y
        scan_kernel<<<(BBATCH * DIN * NST) / 256, 256, 0, stream>>>(
            u, delta, dblb, A_log + (size_t)i * DIN * NST, Dp + (size_t)i * DIN, xz, yh);
        // h = y @ out_proj^T                       (f32 for pool + f16 for next layer)
        launch_gemm(yh, op_wh + (size_t)i * HDIM * DIN, nullptr,
                    hbuf, hbufh, BL, HDIM, DIN, DIN, 0, stream);
    }

    // max over time + FC -> (B,1)
    final_kernel<<<BBATCH, 256, 0, stream>>>(hbuf, fc_w, fc_b, out);
}
